// VisionTransformer_12773232739144
// MI455X (gfx1250) — compile-verified, matrix operand order checked by reading, not run on hardware
//
#include <hip/hip_runtime.h>
#include <math.h>

// ---------------- model constants ----------------
#define BB 2
#define DD 768
#define HHD 12
#define LLAYERS 4
#define RR 16
#define SSS 98
#define NNTOK 1568          // R * S
#define SEQ1 99             // S + 1
#define SCALE_ATTN 0.125f   // 64^-0.5

typedef __bf16 bf16x;
typedef __attribute__((ext_vector_type(16))) __bf16 v16bf;
typedef __attribute__((ext_vector_type(8)))  float  v8f;

// ================= WMMA GEMM (bf16 in, f32 acc) =================
// C[M,N] = A[M,K] * W[N,K]^T ; A,W bf16 row-major; output f32 or bf16.
// 128x64 block tile, 8 waves as 4(M)x2(N), each wave 32x32 = 4 WMMA/K-step.
// Double-buffered LDS filled with GLOBAL_LOAD_ASYNC_TO_LDS_B128 (ASYNCcnt).
#define BM 128
#define BN 64
#define BK 32
#define LST 40   // bf16 elems per LDS row (80B rows: aligned 16B frags, conflict-free)

#define FLAG_GELU 1
#define FLAG_OUTBF16 2

#define ASYNC_B128(ldsoff, gaddr) \
  asm volatile("global_load_async_to_lds_b128 %0, %1, off" \
               :: "v"(ldsoff), "v"(gaddr) : "memory")

__global__ __launch_bounds__(256) void gemm_wmma(
    const bf16x* __restrict__ A, const bf16x* __restrict__ W,
    void* __restrict__ Cout, const float* __restrict__ bias,
    const float* __restrict__ resid,
    int M, int N, int K, int flags)
{
  __shared__ __align__(16) bf16x lA[2][BM * LST];
  __shared__ __align__(16) bf16x lB[2][BN * LST];

  const int tid  = threadIdx.x;
  const int wave = tid >> 5;
  const int lane = tid & 31;
  const int wm   = wave >> 1;       // 0..3 : 32-row slice
  const int wn   = wave & 1;        // 0..1 : 32-col slice
  const int m0   = blockIdx.y * BM;
  const int n0   = blockIdx.x * BN;
  const int khalf = lane >> 4;
  const int l15   = lane & 15;

  // async staging map: 16B segments (8 bf16). A: 128 rows x 4 segs = 512 (2/thread).
  // B: 64 rows x 4 segs = 256 (1/thread). OOB rows clamped (garbage guarded in epilogue).
  const int ra0 = tid >> 1;                    // 0..127  (segments tid*2, tid*2+1 -> same row!)
  const int ca0 = (tid & 1) * 16;              // elem col 0 or 16 (two 16B segs? no: 8 elems)
  // Use explicit seg ids instead:
  const int sA0 = tid;                         // 0..255
  const int sA1 = tid + 256;                   // 256..511
  const int rowA0 = sA0 >> 2, colA0 = (sA0 & 3) * 8;
  const int rowA1 = sA1 >> 2, colA1 = (sA1 & 3) * 8;
  const int rowB  = tid >> 2, colB  = (tid & 3) * 8;
  (void)ra0; (void)ca0;

  const int grA0 = (m0 + rowA0 < M) ? (m0 + rowA0) : (M - 1);
  const int grA1 = (m0 + rowA1 < M) ? (m0 + rowA1) : (M - 1);
  const int grB  = (n0 + rowB  < N) ? (n0 + rowB)  : (N - 1);

  unsigned long long gA0 = (unsigned long long)(uintptr_t)A + 2ull * ((size_t)grA0 * K + colA0);
  unsigned long long gA1 = (unsigned long long)(uintptr_t)A + 2ull * ((size_t)grA1 * K + colA1);
  unsigned long long gB  = (unsigned long long)(uintptr_t)W + 2ull * ((size_t)grB  * K + colB);

  const unsigned dA0[2] = { (unsigned)(uintptr_t)&lA[0][rowA0 * LST + colA0],
                            (unsigned)(uintptr_t)&lA[1][rowA0 * LST + colA0] };
  const unsigned dA1[2] = { (unsigned)(uintptr_t)&lA[0][rowA1 * LST + colA1],
                            (unsigned)(uintptr_t)&lA[1][rowA1 * LST + colA1] };
  const unsigned dB[2]  = { (unsigned)(uintptr_t)&lB[0][rowB * LST + colB],
                            (unsigned)(uintptr_t)&lB[1][rowB * LST + colB] };

  v8f acc00 = {}, acc01 = {}, acc10 = {}, acc11 = {};

  // prime buffer 0
  ASYNC_B128(dA0[0], gA0);
  ASYNC_B128(dA1[0], gA1);
  ASYNC_B128(dB[0],  gB);

  const int nk = K >> 5;
  for (int kb = 0; kb < nk; kb++) {
    asm volatile("s_wait_asynccnt 0" ::: "memory");
    __syncthreads();
    if (kb + 1 < nk) {
      gA0 += 64; gA1 += 64; gB += 64;   // advance 32 bf16 = 64B
      const int nb = (kb + 1) & 1;
      ASYNC_B128(dA0[nb], gA0);
      ASYNC_B128(dA1[nb], gA1);
      ASYNC_B128(dB[nb],  gB);
    }
    const int cur = kb & 1;
    const bf16x* pa0 = &lA[cur][(wm * 32 + l15) * LST];
    const bf16x* pa1 = &lA[cur][(wm * 32 + 16 + l15) * LST];
    const bf16x* pb0 = &lB[cur][(wn * 32 + l15) * LST];
    const bf16x* pb1 = &lB[cur][(wn * 32 + 16 + l15) * LST];
    v16bf fa0, fa1, fb0, fb1;
#pragma unroll
    for (int i = 0; i < 8; i++) {
      fa0[i]     = pa0[khalf * 8 + i];
      fa0[8 + i] = pa0[16 + khalf * 8 + i];
      fa1[i]     = pa1[khalf * 8 + i];
      fa1[8 + i] = pa1[16 + khalf * 8 + i];
      fb0[i]     = pb0[khalf * 8 + i];
      fb0[8 + i] = pb0[16 + khalf * 8 + i];
      fb1[i]     = pb1[khalf * 8 + i];
      fb1[8 + i] = pb1[16 + khalf * 8 + i];
    }
    acc00 = __builtin_amdgcn_wmma_f32_16x16x32_bf16(false, fa0, false, fb0, (short)0, acc00, false, false);
    acc01 = __builtin_amdgcn_wmma_f32_16x16x32_bf16(false, fa0, false, fb1, (short)0, acc01, false, false);
    acc10 = __builtin_amdgcn_wmma_f32_16x16x32_bf16(false, fa1, false, fb0, (short)0, acc10, false, false);
    acc11 = __builtin_amdgcn_wmma_f32_16x16x32_bf16(false, fa1, false, fb1, (short)0, acc11, false, false);
  }

  // epilogue
#pragma unroll
  for (int j = 0; j < 8; j++) {
    const int r0 = m0 + wm * 32 + (khalf ? 8 + j : j);
    const int r1 = r0 + 16;
    const int c0 = n0 + wn * 32 + l15;
    const int c1 = c0 + 16;
    float vals[4] = { acc00[j], acc01[j], acc10[j], acc11[j] };
    int rows[4] = { r0, r0, r1, r1 };
    int cols[4] = { c0, c1, c0, c1 };
#pragma unroll
    for (int t = 0; t < 4; t++) {
      int row = rows[t], col = cols[t];
      if (row >= M || col >= N) continue;
      float v = vals[t];
      if (bias) v += bias[col];
      if (flags & FLAG_GELU) v = 0.5f * v * (1.0f + erff(v * 0.70710678118654752f));
      size_t o = (size_t)row * N + col;
      if (flags & FLAG_OUTBF16) {
        ((bf16x*)Cout)[o] = (bf16x)v;
      } else {
        if (resid) v += resid[o];
        ((float*)Cout)[o] = v;
      }
    }
  }
}

// ================= LayerNorm (row of 768) -> bf16 =================
__global__ __launch_bounds__(256) void ln_kernel(
    const float* __restrict__ in, bf16x* __restrict__ out,
    const float* __restrict__ g, const float* __restrict__ b)
{
  __shared__ float red[256];
  const int row = blockIdx.x, tid = threadIdx.x;
  const float* x = in + (size_t)row * DD;
  float v0 = x[tid], v1 = x[tid + 256], v2 = x[tid + 512];
  red[tid] = v0 + v1 + v2;
  __syncthreads();
  for (int s = 128; s > 0; s >>= 1) { if (tid < s) red[tid] += red[tid + s]; __syncthreads(); }
  float mu = red[0] * (1.0f / DD);
  __syncthreads();
  float d0 = v0 - mu, d1 = v1 - mu, d2 = v2 - mu;
  red[tid] = d0 * d0 + d1 * d1 + d2 * d2;
  __syncthreads();
  for (int s = 128; s > 0; s >>= 1) { if (tid < s) red[tid] += red[tid + s]; __syncthreads(); }
  float inv = 1.0f / sqrtf(red[0] * (1.0f / DD) + 1e-5f);
  bf16x* o = out + (size_t)row * DD;
  o[tid]       = (bf16x)(d0 * inv * g[tid]       + b[tid]);
  o[tid + 256] = (bf16x)(d1 * inv * g[tid + 256] + b[tid + 256]);
  o[tid + 512] = (bf16x)(d2 * inv * g[tid + 512] + b[tid + 512]);
}

// ================= self-attention (seq <= 99, H=12, DH=64) =================
__global__ __launch_bounds__(128) void attn_self(
    const float* __restrict__ qkv, bf16x* __restrict__ out, int seqlen)
{
  __shared__ float ksh[SEQ1 * 64];
  __shared__ float vsh[SEQ1 * 64];
  const int h  = blockIdx.x % HHD;
  const int sq = blockIdx.x / HHD;
  const size_t base = (size_t)sq * seqlen;
  const int tid = threadIdx.x;
  const int tot = seqlen * 64;
  for (int i = tid; i < tot; i += 128) {
    int s = i >> 6, d = i & 63;
    ksh[i] = qkv[(base + s) * 2304 +  768 + h * 64 + d];
    vsh[i] = qkv[(base + s) * 2304 + 1536 + h * 64 + d];
  }
  __syncthreads();
  if (tid >= seqlen) return;
  float qv[64];
#pragma unroll
  for (int d = 0; d < 64; d++) qv[d] = qkv[(base + tid) * 2304 + h * 64 + d] * SCALE_ATTN;
  float mx = -1e30f;
  for (int j = 0; j < seqlen; j++) {
    float s = 0.f;
#pragma unroll
    for (int d = 0; d < 64; d++) s += qv[d] * ksh[j * 64 + d];
    mx = fmaxf(mx, s);
  }
  float sum = 0.f, acc[64];
#pragma unroll
  for (int d = 0; d < 64; d++) acc[d] = 0.f;
  for (int j = 0; j < seqlen; j++) {
    float s = 0.f;
#pragma unroll
    for (int d = 0; d < 64; d++) s += qv[d] * ksh[j * 64 + d];
    float w = expf(s - mx);
    sum += w;
#pragma unroll
    for (int d = 0; d < 64; d++) acc[d] += w * vsh[j * 64 + d];
  }
  float inv = 1.0f / sum;
#pragma unroll
  for (int d = 0; d < 64; d++)
    out[(base + tid) * DD + h * 64 + d] = (bf16x)(acc[d] * inv);
}

// ================= cross-attention: 3136 queries x 16 keys =================
__global__ __launch_bounds__(256) void attn_cross(
    const bf16x* __restrict__ q, const float* __restrict__ kv,
    bf16x* __restrict__ out)
{
  const int idx = blockIdx.x * 256 + threadIdx.x;
  const int total = BB * NNTOK * HHD;
  if (idx >= total) return;
  const int h = idx % HHD;
  const int t = (idx / HHD) % NNTOK;
  const int b = idx / (HHD * NNTOK);
  const bf16x* qp = q + ((size_t)b * NNTOK + t) * DD + h * 64;
  float qv[64];
#pragma unroll
  for (int d = 0; d < 64; d++) qv[d] = (float)qp[d] * SCALE_ATTN;
  float sc[16]; float mx = -1e30f;
  for (int j = 0; j < 16; j++) {
    const float* kp = kv + ((size_t)b * 16 + j) * 1536 + h * 64;
    float s = 0.f;
#pragma unroll
    for (int d = 0; d < 64; d++) s += qv[d] * kp[d];
    sc[j] = s; mx = fmaxf(mx, s);
  }
  float sum = 0.f;
  for (int j = 0; j < 16; j++) { sc[j] = expf(sc[j] - mx); sum += sc[j]; }
  float inv = 1.0f / sum;
  bf16x* op = out + ((size_t)b * NNTOK + t) * DD + h * 64;
#pragma unroll
  for (int d = 0; d < 64; d++) {
    float a = 0.f;
    for (int j = 0; j < 16; j++)
      a += sc[j] * kv[((size_t)b * 16 + j) * 1536 + 768 + h * 64 + d];
    op[d] = (bf16x)(a * inv);
  }
}

// ================= element-wise helpers =================
__global__ void cvt_kernel(const float* __restrict__ s, bf16x* __restrict__ d, long n)
{
  long i = (long)blockIdx.x * 256 + threadIdx.x;
  if (i < n) d[i] = (bf16x)s[i];
}

__global__ void im2col_kernel(const float* __restrict__ video, bf16x* __restrict__ patches)
{
  long idx = (long)blockIdx.x * 256 + threadIdx.x;
  const long total = (long)BB * NNTOK * 1536;
  if (idx >= total) return;
  int kk = (int)(idx % 1536);
  long token = idx / 1536;
  int n = (int)(token % NNTOK);
  int b = (int)(token / NNTOK);
  int dt = n / 196, hh = (n / 14) % 14, ww = n % 14;
  int c = kk >> 9, r = kk & 511, zd = r >> 8, r2 = r & 255, zh = r2 >> 4, zw = r2 & 15;
  long vidx = ((((long)b * 3 + c) * 16 + (dt * 2 + zd)) * 224 + (hh * 16 + zh)) * 224 + (ww * 16 + zw);
  patches[idx] = (bf16x)video[vidx];
}

__global__ void addpos_kernel(float* __restrict__ x)
{
  long idx = (long)blockIdx.x * 256 + threadIdx.x;
  const long total = (long)BB * NNTOK * DD;
  if (idx >= total) return;
  int d = (int)(idx % DD);
  int n = (int)((idx / DD) % NNTOK);
  float e = powf(10000.0f, (2.0f * (float)(d / 2)) / (float)DD);
  float ang = (float)n / e;
  x[idx] += (d & 1) ? cosf(ang) : sinf(ang);
}

__global__ void initm_kernel(const float* __restrict__ rt, float* __restrict__ m)
{
  int idx = blockIdx.x * 256 + threadIdx.x;
  const int total = BB * RR * DD;
  if (idx >= total) return;
  m[idx] = rt[idx % (RR * DD)];
}

__global__ void gather_z(const float* __restrict__ x, const float* __restrict__ m,
                         float* __restrict__ z)
{
  long idx = (long)blockIdx.x * 256 + threadIdx.x;
  const long total = (long)BB * RR * SEQ1 * DD;
  if (idx >= total) return;
  int d  = (int)(idx % DD);
  int s1 = (int)((idx / DD) % SEQ1);
  int bR = (int)(idx / ((long)DD * SEQ1));
  int b = bR >> 4, r = bR & 15;
  if (s1 == 0) {
    z[idx] = m[((size_t)b * RR + r) * DD + d];
  } else {
    int s = s1 - 1;
    int nt = r >> 2, nh = (r >> 1) & 1, nw = r & 1;
    int rt = s / 49, rh = (s / 7) % 7, rw = s % 7;
    int n = ((((nt * 2 + rt) * 2 + nh) * 7 + rh) * 2 + nw) * 7 + rw;
    z[idx] = x[((size_t)b * NNTOK + n) * DD + d];
  }
}

__global__ void extract_m2(const float* __restrict__ z, float* __restrict__ m2)
{
  int idx = blockIdx.x * 256 + threadIdx.x;
  const int total = BB * RR * DD;
  if (idx >= total) return;
  int d = idx % DD, bR = idx / DD;
  m2[idx] = z[(size_t)bR * SEQ1 * DD + d];
}

__global__ void scatter_local(const float* __restrict__ z, float* __restrict__ x)
{
  long idx = (long)blockIdx.x * 256 + threadIdx.x;
  const long total = (long)BB * RR * SSS * DD;
  if (idx >= total) return;
  int d = (int)(idx % DD);
  int s = (int)((idx / DD) % SSS);
  int bR = (int)(idx / ((long)DD * SSS));
  int b = bR >> 4, r = bR & 15;
  x[((size_t)b * NNTOK + r * SSS + s) * DD + d] =
      z[(size_t)bR * SEQ1 * DD + (size_t)(s + 1) * DD + d];
}

__global__ void pool_kernel(const float* __restrict__ x, float* __restrict__ out)
{
  int b = blockIdx.x, tid = threadIdx.x;
  for (int d = tid; d < DD; d += 256) {
    float s = 0.f;
    for (int n = 0; n < NNTOK; n++) s += x[((size_t)b * NNTOK + n) * DD + d];
    out[b * DD + d] = s * (1.0f / NNTOK);
  }
}

// ================= host orchestration =================
static inline int cdiv(long a, int b) { return (int)((a + b - 1) / b); }

extern "C" void kernel_launch(void* const* d_in, const int* in_sizes, int n_in,
                              void* d_out, int out_size, void* d_ws, size_t ws_size,
                              hipStream_t stream)
{
  const float* video   = (const float*)d_in[0];
  const float* conv_w  = (const float*)d_in[1];
  const float* conv_b  = (const float*)d_in[2];
  const float* regtok  = (const float*)d_in[3];
  const float* ln1_s   = (const float*)d_in[4];
  const float* ln1_b   = (const float*)d_in[5];
  const float* ln2_s   = (const float*)d_in[6];
  const float* ln2_b   = (const float*)d_in[7];
  const float* ln3a_s  = (const float*)d_in[8];
  const float* ln3a_b  = (const float*)d_in[9];
  const float* ln3b_s  = (const float*)d_in[10];
  const float* ln3b_b  = (const float*)d_in[11];
  const float* mlpln_s = (const float*)d_in[12];
  const float* mlpln_b = (const float*)d_in[13];
  const float* mmln_s  = (const float*)d_in[14];
  const float* mmln_b  = (const float*)d_in[15];
  const float* a1_qkv  = (const float*)d_in[16];
  const float* a1_pw   = (const float*)d_in[17];
  const float* a1_pb   = (const float*)d_in[18];
  const float* a2_qkv  = (const float*)d_in[19];
  const float* a2_pw   = (const float*)d_in[20];
  const float* a2_pb   = (const float*)d_in[21];
  const float* cq_w    = (const float*)d_in[22];
  const float* ckv_w   = (const float*)d_in[23];
  const float* cp_w    = (const float*)d_in[24];
  const float* cp_b    = (const float*)d_in[25];
  const float* f1_w    = (const float*)d_in[26];
  const float* f1_b    = (const float*)d_in[27];
  const float* f2_w    = (const float*)d_in[28];
  const float* f2_b    = (const float*)d_in[29];
  const float* g1_w    = (const float*)d_in[30];
  const float* g1_b    = (const float*)d_in[31];
  const float* g2_w    = (const float*)d_in[32];
  const float* g2_b    = (const float*)d_in[33];
  const float* fcn_s   = (const float*)d_in[34];
  const float* fcn_b   = (const float*)d_in[35];
  const float* head_w  = (const float*)d_in[36];
  const float* head_b  = (const float*)d_in[37];
  float* outp = (float*)d_out;

  // ---- workspace layout: f32 region then bf16 region ----
  float* ws = (float*)d_ws;
  float* xbuf   = ws;                     // 2,408,448
  float* zbuf   = xbuf + 2408448;         // 2,433,024
  float* qkvb   = zbuf + 2433024;         // 7,299,072
  float* kvbuf  = qkvb + 7299072;         // 49,152
  float* mbuf   = kvbuf + 49152;          // 24,576
  float* m2buf  = mbuf + 24576;           // 24,576
  float* pooled = m2buf + 24576;          // 1,536

  bf16x* lnb    = (bf16x*)(pooled + 1536);// 2,433,024
  bf16x* attno  = lnb + 2433024;          // 2,433,024
  bf16x* qbuf   = attno + 2433024;        // 2,408,448
  bf16x* hidden = qbuf + 2408448;         // 9,633,792 (also im2col patches)
  bf16x* mnb    = hidden + 9633792;       // 24,576
  bf16x* mhid   = mnb + 24576;            // 98,304
  bf16x* plnb   = mhid + 98304;           // 1,536
  // bf16 weights
  bf16x* convB  = plnb + 1536;            // 1,179,648
  bf16x* a1qB   = convB + 1179648;        // 7,077,888
  bf16x* a1pB   = a1qB + 7077888;         // 2,359,296
  bf16x* a2qB   = a1pB + 2359296;         // 7,077,888
  bf16x* a2pB   = a2qB + 7077888;         // 2,359,296
  bf16x* cqB    = a2pB + 2359296;         // 2,359,296
  bf16x* ckvB   = cqB + 2359296;          // 4,718,592
  bf16x* cpB    = ckvB + 4718592;         // 2,359,296
  bf16x* f1B    = cpB + 2359296;          // 9,437,184
  bf16x* f2B    = f1B + 9437184;          // 9,437,184
  bf16x* g1B    = f2B + 9437184;          // 9,437,184
  bf16x* g2B    = g1B + 9437184;          // 9,437,184
  bf16x* headB  = g2B + 9437184;          // 768,000

  const int M_ALL = BB * RR * SEQ1;   // 3168
  const int M_LOC = BB * NNTOK;       // 3136
  const int M_REG = BB * RR;          // 32

  auto cvt = [&](const float* s, bf16x* d, long n) {
    cvt_kernel<<<cdiv(n, 256), 256, 0, stream>>>(s, d, n);
  };
  auto gemm = [&](const bf16x* A, const bf16x* W, void* Cp,
                  const float* bias, const float* res,
                  int M, int N, int K, int flags) {
    dim3 g((N + BN - 1) / BN, (M + BM - 1) / BM);
    gemm_wmma<<<g, 256, 0, stream>>>(A, W, Cp, bias, res, M, N, K, flags);
  };

  // ---- pre-convert weights to bf16 (once per launch; ~400MB @ 23.3TB/s) ----
  cvt(conv_w, convB, 1179648);
  cvt(a1_qkv, a1qB, 7077888);
  cvt(a1_pw,  a1pB, 2359296);
  cvt(a2_qkv, a2qB, 7077888);
  cvt(a2_pw,  a2pB, 2359296);
  cvt(cq_w,   cqB,  2359296);
  cvt(ckv_w,  ckvB, 4718592);
  cvt(cp_w,   cpB,  2359296);
  cvt(f1_w,   f1B,  9437184);
  cvt(f2_w,   f2B,  9437184);
  cvt(g1_w,   g1B,  9437184);
  cvt(g2_w,   g2B,  9437184);
  cvt(head_w, headB, 768000);

  // ---- patch embed ----
  im2col_kernel<<<cdiv((long)M_LOC * 1536, 256), 256, 0, stream>>>(video, hidden);
  gemm(hidden, convB, xbuf, conv_b, nullptr, M_LOC, DD, 1536, 0);
  addpos_kernel<<<cdiv((long)M_LOC * DD, 256), 256, 0, stream>>>(xbuf);
  initm_kernel<<<cdiv(BB * RR * DD, 256), 256, 0, stream>>>(regtok, mbuf);

  for (int l = 0; l < LLAYERS; l++) {
    const size_t oD  = (size_t)l * DD;
    const size_t oQ  = (size_t)l * 3 * DD * DD;
    const size_t oP  = (size_t)l * DD * DD;
    const size_t oKV = (size_t)l * 2 * DD * DD;
    const size_t oF1 = (size_t)l * 4 * DD * DD;
    const size_t oB1 = (size_t)l * 4 * DD;

    // (a) local MHA over 32 x 99
    gather_z<<<cdiv((long)M_ALL * DD, 256), 256, 0, stream>>>(xbuf, mbuf, zbuf);
    ln_kernel<<<M_ALL, 256, 0, stream>>>(zbuf, lnb, ln1_s + oD, ln1_b + oD);
    gemm(lnb, a1qB + oQ, qkvb, nullptr, nullptr, M_ALL, 3 * DD, DD, 0);
    attn_self<<<M_REG * HHD, 128, 0, stream>>>(qkvb, attno, SEQ1);
    gemm(attno, a1pB + oP, zbuf, a1_pb + oD, zbuf, M_ALL, DD, DD, 0);

    // (b) region MHA over 2 x 16
    extract_m2<<<cdiv(M_REG * DD, 256), 256, 0, stream>>>(zbuf, m2buf);
    ln_kernel<<<M_REG, 256, 0, stream>>>(m2buf, mnb, ln2_s + oD, ln2_b + oD);
    gemm(mnb, a2qB + oQ, qkvb, nullptr, nullptr, M_REG, 3 * DD, DD, 0);
    attn_self<<<BB * HHD, 128, 0, stream>>>(qkvb, attno, RR);
    gemm(attno, a2pB + oP, m2buf, a2_pb + oD, m2buf, M_REG, DD, DD, 0);

    // (c) cross-attention
    scatter_local<<<cdiv((long)BB * RR * SSS * DD, 256), 256, 0, stream>>>(zbuf, xbuf);
    ln_kernel<<<M_LOC, 256, 0, stream>>>(xbuf, lnb, ln3a_s + oD, ln3a_b + oD);
    gemm(lnb, cqB + oP, qbuf, nullptr, nullptr, M_LOC, DD, DD, FLAG_OUTBF16);
    ln_kernel<<<M_REG, 256, 0, stream>>>(m2buf, mnb, ln3b_s + oD, ln3b_b + oD);
    gemm(mnb, ckvB + oKV, kvbuf, nullptr, nullptr, M_REG, 2 * DD, DD, 0);
    attn_cross<<<cdiv(BB * NNTOK * HHD, 256), 256, 0, stream>>>(qbuf, kvbuf, attno);
    gemm(attno, cpB + oP, xbuf, cp_b + oD, xbuf, M_LOC, DD, DD, 0);

    // (d) local MLP
    ln_kernel<<<M_LOC, 256, 0, stream>>>(xbuf, lnb, mlpln_s + oD, mlpln_b + oD);
    gemm(lnb, f1B + oF1, hidden, f1_b + oB1, nullptr, M_LOC, 4 * DD, DD, FLAG_GELU | FLAG_OUTBF16);
    gemm(hidden, f2B + oF1, xbuf, f2_b + oD, xbuf, M_LOC, DD, 4 * DD, 0);

    // (e) region MLP -> m
    ln_kernel<<<M_REG, 256, 0, stream>>>(m2buf, mnb, mmln_s + oD, mmln_b + oD);
    gemm(mnb, g1B + oF1, mhid, g1_b + oB1, nullptr, M_REG, 4 * DD, DD, FLAG_GELU | FLAG_OUTBF16);
    gemm(mhid, g2B + oF1, mbuf, g2_b + oD, m2buf, M_REG, DD, 4 * DD, 0);
  }

  // ---- head ----
  pool_kernel<<<BB, 256, 0, stream>>>(xbuf, pooled);
  ln_kernel<<<BB, 256, 0, stream>>>(pooled, plnb, fcn_s, fcn_b);
  gemm(plnb, headB, outp, head_b, nullptr, BB, 1000, DD, 0);
}
